// TemporalMambaBlock_17781164605861
// MI455X (gfx1250) — matrix operand order repaired and checked by reading, not compiled, not run on hardware
//
#include <hip/hip_runtime.h>
#include <hip/hip_bf16.h>

// ---------------------------------------------------------------------------
// Mamba block on gfx1250: 2x2 register-blocked WMMA f16 GEMMs (f32 accumulate)
// + channel-parallel selective scan with LDS staging of shared B/C vectors.
// 2x2 chosen so accumulators+fragments (~90 VGPRs) stay under the allocator
// budget -- the 2x4 variant spilled accumulators to scratch in the k-loop.
// ---------------------------------------------------------------------------

typedef __attribute__((ext_vector_type(16))) _Float16 v16h;
typedef __attribute__((ext_vector_type(8)))  _Float16 half8;
typedef __attribute__((ext_vector_type(8)))  float    v8f;

#define DIMC    1024
#define LSEQ    1024
#define BATCH   2
#define DSTATE  16
#define DCONV   4
#define EDC     2048            // EXPAND * DIM
#define DTRANK  64
#define NROWS   (BATCH * LSEQ)  // 2048 token rows

// ---------------------------------------------------------------------------
// WMMA fragment loaders (ISA 7.12.2 layouts).
//  A (16x32 f16): lanes 0-15 hold K-octets {0-7,16-23}; lanes 16-31 {8-15,24-31}
//    -> caller pre-offsets the row pointer by ((lane&16)?8:0).
//  B from transposed weights WT (N x K row-major): same K-octets as A
//    -> caller pre-offsets the column pointer by ((lane&16)?8:0).
// ---------------------------------------------------------------------------
__device__ __forceinline__ v16h load_a_frag(const _Float16* p) {
    half8 lo = *(const half8*)p;
    half8 hi = *(const half8*)(p + 16);
    v16h r;
#pragma unroll
    for (int i = 0; i < 8; ++i) { r[i] = lo[i]; r[i + 8] = hi[i]; }
    return r;
}
__device__ __forceinline__ v16h load_b_frag(const _Float16* p) {
    half8 lo = *(const half8*)p;
    half8 hi = *(const half8*)(p + 16);
    v16h r;
#pragma unroll
    for (int i = 0; i < 8; ++i) { r[i] = lo[i]; r[i + 8] = hi[i]; }
    return r;
}

// ---------------------------------------------------------------------------
// Weight convert + transpose: W (K x N, f32 row-major) -> WT (N x K, f16).
// ---------------------------------------------------------------------------
__global__ void k_cvt_transpose(const float* __restrict__ W, _Float16* __restrict__ WT,
                                int K, int N) {
    int idx = blockIdx.x * blockDim.x + threadIdx.x;
    if (idx >= K * N) return;
    int k = idx / N, n = idx % N;
    WT[(size_t)n * K + k] = (_Float16)W[idx];
}

// ---------------------------------------------------------------------------
// RMSNorm: one block per token row, write f16 activations.
// ---------------------------------------------------------------------------
__global__ void k_rmsnorm(const float* __restrict__ x, const float* __restrict__ w,
                          _Float16* __restrict__ xn) {
    __shared__ float red[256];
    int row = blockIdx.x;
    const float* xr = x + (size_t)row * DIMC;
    float ss = 0.f;
    for (int c = threadIdx.x; c < DIMC; c += 256) { float v = xr[c]; ss += v * v; }
    red[threadIdx.x] = ss;
    __syncthreads();
    for (int s = 128; s > 0; s >>= 1) {
        if (threadIdx.x < s) red[threadIdx.x] += red[threadIdx.x + s];
        __syncthreads();
    }
    float scale = rsqrtf(red[0] / (float)DIMC + 1e-5f);
    for (int c = threadIdx.x; c < DIMC; c += 256)
        xn[(size_t)row * DIMC + c] = (_Float16)(xr[c] * scale * w[c]);
}

// ---------------------------------------------------------------------------
// 2x2 register-blocked WMMA GEMM: each wave computes a 32x32 block (4 tiles
// of 16x16).  Per k-step: 2 A-frags + 2 B-frags -> 4 wmma (2 B128 loads/wmma,
// half the L2 traffic of the 1-tile version, no scratch spills).
// Requires M % 32 == 0, N % 32 == 0, wave-block count % 8 == 0.
// ---------------------------------------------------------------------------
__global__ void __launch_bounds__(256)
k_wmma_gemm_blk(const _Float16* __restrict__ A,
                const _Float16* __restrict__ BT,
                const float* __restrict__ resid,
                float* __restrict__ Dst,
                int M, int N, int K) {
    const int lane = threadIdx.x & 31;
    const int wave = threadIdx.x >> 5;
    const int blks_n = N >> 5;                   // 32-wide blocks
    const int blk = blockIdx.x * 8 + wave;
    const int tm0 = (blk / blks_n) * 32;
    const int tn0 = (blk % blks_n) * 32;

    const int aoff = (lane & 16) ? 8 : 0;
    const int boff = (lane & 16) ? 8 : 0;
    const _Float16* Ap0 = A + (size_t)(tm0 + (lane & 15)) * K + aoff;
    const _Float16* Ap1 = Ap0 + (size_t)16 * K;
    const _Float16* Bp0 = BT + (size_t)(tn0 + (lane & 15)) * K + boff;
    const _Float16* Bp1 = Bp0 + (size_t)16 * K;

    v8f c00 = {}, c01 = {}, c10 = {}, c11 = {};

    for (int k0 = 0; k0 < K; k0 += 32) {
        v16h a0 = load_a_frag(Ap0 + k0);
        v16h a1 = load_a_frag(Ap1 + k0);
        v16h b0 = load_b_frag(Bp0 + k0);
        v16h b1 = load_b_frag(Bp1 + k0);
        c00 = __builtin_amdgcn_wmma_f32_16x16x32_f16(false, a0, false, b0, (short)0, c00, false, false);
        c01 = __builtin_amdgcn_wmma_f32_16x16x32_f16(false, a0, false, b1, (short)0, c01, false, false);
        c10 = __builtin_amdgcn_wmma_f32_16x16x32_f16(false, a1, false, b0, (short)0, c10, false, false);
        c11 = __builtin_amdgcn_wmma_f32_16x16x32_f16(false, a1, false, b1, (short)0, c11, false, false);
    }

    const int nb = lane & 15;
    const int rb = (lane & 16) ? 8 : 0;
#pragma unroll
    for (int r = 0; r < 8; ++r) {
        size_t row0 = (size_t)(tm0 + rb + r) * N;
        size_t row1 = (size_t)(tm0 + 16 + rb + r) * N;
        size_t o00 = row0 + tn0 + nb;
        size_t o01 = row0 + tn0 + 16 + nb;
        size_t o10 = row1 + tn0 + nb;
        size_t o11 = row1 + tn0 + 16 + nb;
        float v00 = c00[r], v01 = c01[r], v10 = c10[r], v11 = c11[r];
        if (resid) {
            v00 += resid[o00]; v01 += resid[o01];
            v10 += resid[o10]; v11 += resid[o11];
        }
        Dst[o00] = v00; Dst[o01] = v01;
        Dst[o10] = v10; Dst[o11] = v11;
    }
}

// ---------------------------------------------------------------------------
// Simple WMMA GEMM (1 tile / wave) for the narrow N=96 projection.
// ---------------------------------------------------------------------------
__global__ void __launch_bounds__(256)
k_wmma_gemm(const _Float16* __restrict__ A,
            const _Float16* __restrict__ BT,
            float* __restrict__ Dst,
            int M, int N, int K) {
    const int lane = threadIdx.x & 31;
    const int wave = threadIdx.x >> 5;
    const int tiles_n = N >> 4;
    const int tile = blockIdx.x * 8 + wave;
    const int tm = tile / tiles_n;
    const int tn = tile % tiles_n;

    const _Float16* Ap = A  + (size_t)(tm * 16 + (lane & 15)) * K + ((lane & 16) ? 8 : 0);
    const _Float16* Bp = BT + (size_t)(tn * 16 + (lane & 15)) * K + ((lane & 16) ? 8 : 0);

    v8f c = {};
#pragma unroll 2
    for (int k0 = 0; k0 < K; k0 += 32) {
        v16h av = load_a_frag(Ap + k0);
        v16h bv = load_b_frag(Bp + k0);
        c = __builtin_amdgcn_wmma_f32_16x16x32_f16(false, av, false, bv,
                                                   (short)0, c, false, false);
    }

    const int n = tn * 16 + (lane & 15);
    const int rbase = tm * 16 + ((lane & 16) ? 8 : 0);
#pragma unroll
    for (int r = 0; r < 8; ++r)
        Dst[(size_t)(rbase + r) * N + n] = c[r];
}

// ---------------------------------------------------------------------------
// Causal depthwise conv (taps=4) + bias + SiLU on the xb half of xz -> f16.
// ---------------------------------------------------------------------------
__global__ void k_conv_silu(const float* __restrict__ xz,
                            const float* __restrict__ cw,
                            const float* __restrict__ cb,
                            _Float16* __restrict__ xbc) {
    int idx = blockIdx.x * blockDim.x + threadIdx.x;
    if (idx >= BATCH * LSEQ * EDC) return;
    int e = idx & (EDC - 1);
    int t = (idx >> 11) & (LSEQ - 1);
    int b = idx >> 21;
    float acc = cb[e];
#pragma unroll
    for (int k = 0; k < DCONV; ++k) {
        int ts = t - (DCONV - 1) + k;
        if (ts >= 0)
            acc += cw[e * DCONV + k] * xz[((size_t)(b * LSEQ + ts)) * (2 * EDC) + e];
    }
    float s = acc / (1.f + expf(-acc));   // SiLU
    xbc[idx] = (_Float16)s;
}

// ---------------------------------------------------------------------------
// Pack d_r (first 64 cols of dbl) to f16 for GEMM3.
// ---------------------------------------------------------------------------
__global__ void k_pack_dr(const float* __restrict__ dbl, _Float16* __restrict__ dr) {
    int idx = blockIdx.x * blockDim.x + threadIdx.x;
    if (idx >= NROWS * DTRANK) return;
    int row = idx / DTRANK, c = idx % DTRANK;
    dr[idx] = (_Float16)dbl[(size_t)row * (DTRANK + 2 * DSTATE) + c];
}

// ---------------------------------------------------------------------------
// delta = softplus(delta_raw + b_dt), in place.
// ---------------------------------------------------------------------------
__global__ void k_softplus_bias(float* __restrict__ delta, const float* __restrict__ bdt) {
    int idx = blockIdx.x * blockDim.x + threadIdx.x;
    if (idx >= NROWS * EDC) return;
    float v = delta[idx] + bdt[idx & (EDC - 1)];
    delta[idx] = (v > 20.f) ? v : log1pf(expf(v));
}

// ---------------------------------------------------------------------------
// Selective scan, channel-parallel: 16 blocks x 256 lanes = 4096 (b,e) chans.
// Each lane keeps h[16] in VGPRs; shared B_t/C_t staged via LDS per 16 steps.
// Prefetches the next chunk's dbl rows while computing the current chunk.
// Fuses the D*x skip term and the silu(z) gate; writes gated y as f16.
// ---------------------------------------------------------------------------
__global__ void k_scan(const float* __restrict__ dbl,
                       const float* __restrict__ delta,
                       const _Float16* __restrict__ xbc,
                       const float* __restrict__ xz,
                       const float* __restrict__ A_log,
                       const float* __restrict__ Dp,
                       _Float16* __restrict__ yg) {
    __shared__ float sB[16][DSTATE];
    __shared__ float sC[16][DSTATE];
    const int b = blockIdx.x >> 3;
    const int e = ((blockIdx.x & 7) << 8) + threadIdx.x;

    float A[DSTATE], h[DSTATE];
#pragma unroll
    for (int n = 0; n < DSTATE; ++n) {
        A[n] = -expf(A_log[e * DSTATE + n]);
        h[n] = 0.f;
    }
    const float De = Dp[e];

    for (int t0 = 0; t0 < LSEQ; t0 += 16) {
        __syncthreads();
        for (int i = threadIdx.x; i < 16 * 2 * DSTATE; i += 256) {
            int tt = i >> 5, j = i & 31;
            float v = dbl[((size_t)(b * LSEQ + t0 + tt)) * (DTRANK + 2 * DSTATE) + DTRANK + j];
            if (j < DSTATE) sB[tt][j] = v;
            else            sC[tt][j - DSTATE] = v;
        }
        // Prefetch next chunk's shared rows into cache while we compute.
        if (t0 + 16 < LSEQ) {
            for (int i = threadIdx.x; i < 16 * 2 * DSTATE; i += 256) {
                int tt = i >> 5, j = i & 31;
                __builtin_prefetch(
                    &dbl[((size_t)(b * LSEQ + t0 + 16 + tt)) * (DTRANK + 2 * DSTATE) + DTRANK + j],
                    0, 3);
            }
        }
        __syncthreads();
        for (int tt = 0; tt < 16; ++tt) {
            const size_t rowE = ((size_t)(b * LSEQ + t0 + tt)) * EDC + e;
            const float d  = delta[rowE];
            const float xv = (float)xbc[rowE];
            const float dx = d * xv;
            float y = De * xv;
#pragma unroll
            for (int n = 0; n < DSTATE; ++n) {
                float dA = expf(d * A[n]);
                h[n] = dA * h[n] + dx * sB[tt][n];
                y += h[n] * sC[tt][n];
            }
            const float z = xz[((size_t)(b * LSEQ + t0 + tt)) * (2 * EDC) + EDC + e];
            const float g = z / (1.f + expf(-z));
            yg[rowE] = (_Float16)(y * g);
        }
    }
}

// ---------------------------------------------------------------------------
// Host-side pipeline.
// ---------------------------------------------------------------------------
extern "C" void kernel_launch(void* const* d_in, const int* in_sizes, int n_in,
                              void* d_out, int out_size, void* d_ws, size_t ws_size,
                              hipStream_t stream) {
    const float* x      = (const float*)d_in[0];
    const float* W_in   = (const float*)d_in[1];
    const float* conv_w = (const float*)d_in[2];
    const float* conv_b = (const float*)d_in[3];
    const float* W_x    = (const float*)d_in[4];
    const float* W_dt   = (const float*)d_in[5];
    const float* b_dt   = (const float*)d_in[6];
    const float* A_log  = (const float*)d_in[7];
    const float* Dvec   = (const float*)d_in[8];
    const float* W_out  = (const float*)d_in[9];
    const float* rms_w  = (const float*)d_in[10];
    float* out = (float*)d_out;

    char* ws = (char*)d_ws;
    size_t off = 0;
    auto alloc = [&](size_t bytes) { void* p = ws + off; off = (off + bytes + 255) & ~(size_t)255; return p; };

    _Float16* xn_f16   = (_Float16*)alloc((size_t)NROWS * DIMC * 2);        // 4 MB
    _Float16* WinT     = (_Float16*)alloc((size_t)(2 * EDC) * DIMC * 2);    // 8 MB (4096 x 1024)
    float*    xz       = (float*)   alloc((size_t)NROWS * 2 * EDC * 4);     // 32 MB
    _Float16* xbc_f16  = (_Float16*)alloc((size_t)NROWS * EDC * 2);         // 8 MB
    _Float16* WxT      = (_Float16*)alloc((size_t)(DTRANK + 2 * DSTATE) * EDC * 2);
    float*    dbl      = (float*)   alloc((size_t)NROWS * (DTRANK + 2 * DSTATE) * 4);
    _Float16* dr_f16   = (_Float16*)alloc((size_t)NROWS * DTRANK * 2);
    _Float16* WdtT     = (_Float16*)alloc((size_t)EDC * DTRANK * 2);
    float*    delta    = (float*)   alloc((size_t)NROWS * EDC * 4);         // 16 MB
    _Float16* yg_f16   = (_Float16*)alloc((size_t)NROWS * EDC * 2);         // 8 MB
    _Float16* WoutT    = (_Float16*)alloc((size_t)DIMC * EDC * 2);          // 4 MB
    (void)ws_size; (void)n_in; (void)in_sizes; (void)out_size;

    const int T = 256;

    // Weight conversions (per-call; deterministic, no caching).
    k_cvt_transpose<<<(DIMC * 2 * EDC + T - 1) / T, T, 0, stream>>>(W_in,  WinT,  DIMC, 2 * EDC);
    k_cvt_transpose<<<(EDC * (DTRANK + 2 * DSTATE) + T - 1) / T, T, 0, stream>>>(W_x, WxT, EDC, DTRANK + 2 * DSTATE);
    k_cvt_transpose<<<(DTRANK * EDC + T - 1) / T, T, 0, stream>>>(W_dt,  WdtT,  DTRANK, EDC);
    k_cvt_transpose<<<(EDC * DIMC + T - 1) / T, T, 0, stream>>>(W_out, WoutT, EDC, DIMC);

    // 1) RMSNorm -> f16 activations.
    k_rmsnorm<<<NROWS, T, 0, stream>>>(x, rms_w, xn_f16);

    // 2) xz = xn @ W_in : 2048 x 4096, K = 1024.  blocks32x32 = 64*128 = 8192
    k_wmma_gemm_blk<<<(64 * 128) / 8, T, 0, stream>>>(xn_f16, WinT, nullptr, xz,
                                                      NROWS, 2 * EDC, DIMC);

    // 3) causal depthwise conv + bias + SiLU on xb half.
    k_conv_silu<<<(BATCH * LSEQ * EDC + T - 1) / T, T, 0, stream>>>(xz, conv_w, conv_b, xbc_f16);

    // 4) dbl = xbc @ W_x : 2048 x 96, K = 2048.   tiles = 128*6 = 768
    k_wmma_gemm<<<(128 * 6) / 8, T, 0, stream>>>(xbc_f16, WxT, dbl,
                                                 NROWS, DTRANK + 2 * DSTATE, EDC);

    // 5) delta_raw = d_r @ W_dt : 2048 x 2048, K = 64.  blocks32x32 = 64*64 = 4096
    k_pack_dr<<<(NROWS * DTRANK + T - 1) / T, T, 0, stream>>>(dbl, dr_f16);
    k_wmma_gemm_blk<<<(64 * 64) / 8, T, 0, stream>>>(dr_f16, WdtT, nullptr, delta,
                                                     NROWS, EDC, DTRANK);

    // 6) delta = softplus(delta_raw + b_dt).
    k_softplus_bias<<<(NROWS * EDC + T - 1) / T, T, 0, stream>>>(delta, b_dt);

    // 7) selective scan + D*x skip + silu(z) gate -> gated y (f16).
    k_scan<<<BATCH * (EDC / 256), T, 0, stream>>>(dbl, delta, xbc_f16, xz,
                                                  A_log, Dvec, yg_f16);

    // 8) out = y_gated @ W_out + x : 2048 x 1024, K = 2048. blocks32x32 = 64*32 = 2048
    k_wmma_gemm_blk<<<(64 * 32) / 8, T, 0, stream>>>(yg_f16, WoutT, x, out,
                                                     NROWS, DIMC, EDC);
}
